// ScaledDotProductAttention_11914239279760
// MI455X (gfx1250) — matrix operand order repaired and checked by reading, not compiled, not run on hardware
//
#include <hip/hip_runtime.h>
#include <hip/hip_bf16.h>
#include <stdint.h>
#include <math.h>

#define B_SZ 64
#define N_SZ 1024
#define D_SZ 512

typedef __bf16 bf16_t;
typedef bf16_t v4bf  __attribute__((ext_vector_type(4)));
typedef bf16_t v8bf  __attribute__((ext_vector_type(8)));
typedef bf16_t v16bf __attribute__((ext_vector_type(16)));
typedef float  v8f   __attribute__((ext_vector_type(8)));

// pointee type expected by the async-to-LDS builtin (from hipcc diagnostic)
typedef __attribute__((__vector_size__(4 * sizeof(int)))) int b128i;

#if defined(__HIP_DEVICE_COMPILE__) && __has_builtin(__builtin_amdgcn_global_load_async_to_lds_b128)
#define HAVE_ASYNC_LDS 1
#else
#define HAVE_ASYNC_LDS 0
#endif

static __device__ __forceinline__ v8f zero8() {
  v8f z = {0.f, 0.f, 0.f, 0.f, 0.f, 0.f, 0.f, 0.f};
  return z;
}

// One 16-byte global -> LDS transfer per lane. Async (ASYNCcnt-tracked, no VGPR
// round trip) when the gfx1250 builtin is available, else synchronous copy.
static __device__ __forceinline__ void async_copy_b128(const bf16_t* g, bf16_t* l) {
#if HAVE_ASYNC_LDS
  __builtin_amdgcn_global_load_async_to_lds_b128(
      (__attribute__((address_space(1))) b128i*)(uintptr_t)g,
      (__attribute__((address_space(3))) b128i*)(unsigned)(uintptr_t)l,
      0, 0);
#else
  *(uint4*)l = *(const uint4*)g;
#endif
}

static __device__ __forceinline__ void wait_async0() {
#if HAVE_ASYNC_LDS
#if __has_builtin(__builtin_amdgcn_s_wait_asynccnt)
  __builtin_amdgcn_s_wait_asynccnt(0);
#else
  asm volatile("s_wait_asynccnt 0x0" ::: "memory");
#endif
#endif
}

// A-operand (16x32 bf16, MxK): lane holds row (lane&15); elements 0..7 = K[8h..8h+7],
// elements 8..15 = K[16+8h..16+8h+7]  (h = lane>>4).  rowp points at row start + k0.
static __device__ __forceinline__ v16bf load_a_tile(const bf16_t* rowp, int h) {
  v8bf lo = *(const v8bf*)(rowp + 8 * h);
  v8bf hi = *(const v8bf*)(rowp + 16 + 8 * h);
  v16bf r;
#pragma unroll
  for (int i = 0; i < 8; ++i) { r[i] = lo[i]; r[i + 8] = hi[i]; }
  return r;
}

// B-operand (32x16 bf16, KxN): lane holds column (lane&15); elements 0..7 = K[8h..8h+7],
// elements 8..15 = K[16+8h..16+8h+7]  (h = lane>>4).
static __device__ __forceinline__ v16bf load_b_tile(const bf16_t* colp, int h) {
  v8bf lo = *(const v8bf*)(colp + 8 * h);
  v8bf hi = *(const v8bf*)(colp + 16 + 8 * h);
  v16bf r;
#pragma unroll
  for (int i = 0; i < 8; ++i) { r[i] = lo[i]; r[i + 8] = hi[i]; }
  return r;
}

#define WMMA_BF16(a, b, c) \
  __builtin_amdgcn_wmma_f32_16x16x32_bf16(false, (a), false, (b), (short)0, (c), false, false)

// ---------------------------------------------------------------------------
// f32 -> bf16 (with optional scale)
// ---------------------------------------------------------------------------
__global__ void __launch_bounds__(256)
prep_kernel(const float* __restrict__ src, bf16_t* __restrict__ dst, float scale, int n4) {
  int i = blockIdx.x * blockDim.x + threadIdx.x;
  if (i >= n4) return;
  float4 v = ((const float4*)src)[i];
  v4bf o;
  o[0] = (bf16_t)(v.x * scale);
  o[1] = (bf16_t)(v.y * scale);
  o[2] = (bf16_t)(v.z * scale);
  o[3] = (bf16_t)(v.w * scale);
  ((v4bf*)dst)[i] = o;
}

// ---------------------------------------------------------------------------
// Attention: 1 block = 4 waves = 64 query rows of one batch.
// Stage 1: S = Qs*Ks^T tiles (double-buffered async K tiles) -> exp -> P + row sums.
// Stage 2: O = P*V, normalize, write bf16 attn.
// ---------------------------------------------------------------------------
__global__ void __launch_bounds__(128)
attn_kernel(const bf16_t* __restrict__ xs, const unsigned char* __restrict__ mask,
            bf16_t* __restrict__ attn) {
  extern __shared__ __align__(16) char smem_raw[];
  bf16_t* Qs   = (bf16_t*)smem_raw;          // [64][512]      64 KB
  bf16_t* Ks   = Qs + 64 * D_SZ;             // [2][32][512]   64 KB (double buffer)
  bf16_t* Pb   = Ks + 2 * 32 * D_SZ;         // [64][1024]    128 KB
  bf16_t* Vt   = Pb + 64 * N_SZ;             // [64][32]        4 KB (transposed V chunk)
  float*  rsum = (float*)(Vt + 64 * 32);     // [64]

  const int tid   = threadIdx.x;
  const int wave  = tid >> 5;
  const int lane  = tid & 31;
  const int n16   = lane & 15;
  const int h     = lane >> 4;
  const int qrow0 = wave * 16;               // this wave's 16-row slice within block

  const int b  = blockIdx.x >> 4;            // 16 blocks per batch (1024/64)
  const int q0 = (blockIdx.x & 15) * 64;
  const bf16_t* xb = xs + (size_t)b * N_SZ * D_SZ;
  const unsigned char* mbase = mask + (size_t)b * N_SZ;

  for (int i = tid; i < 64; i += 128) rsum[i] = 0.0f;

  // async cooperative load of Q block (rows q0..q0+63; 32 x b128 per thread)
#pragma unroll 4
  for (int i = tid * 8; i < 64 * D_SZ; i += 128 * 8)
    async_copy_b128(xb + (size_t)q0 * D_SZ + i, Qs + i);

  // preamble: issue key tile 0 into buffer 0
#pragma unroll
  for (int i = 0; i < 16; ++i) {
    const int off = (i * 128 + tid) * 8;
    async_copy_b128(xb + off, Ks + off);
  }

  // ---------------- Stage 1: scores -> exp -> P ----------------
  float psum[8];
#pragma unroll
  for (int r = 0; r < 8; ++r) psum[r] = 0.0f;

  for (int kt = 0; kt < N_SZ / 32; ++kt) {
    bf16_t* kbuf = Ks + (kt & 1) * (32 * D_SZ);
    wait_async0();          // my async loads for tile kt (and Qs) are in LDS
    __syncthreads();        // everyone's tile kt loaded; buf^1 free of readers
    if (kt + 1 < N_SZ / 32) {
      bf16_t* nbuf = Ks + ((kt + 1) & 1) * (32 * D_SZ);
      const bf16_t* src = xb + (size_t)(kt + 1) * 32 * D_SZ;
#pragma unroll
      for (int i = 0; i < 16; ++i) {
        const int off = (i * 128 + tid) * 8;
        async_copy_b128(src + off, nbuf + off);   // overlaps with compute below
      }
    }

    v8f c0 = zero8();
    v8f c1 = zero8();
#pragma unroll
    for (int d0 = 0; d0 < D_SZ; d0 += 32) {
      v16bf a  = load_a_tile(Qs + (qrow0 + n16) * D_SZ + d0, h);
      // B columns of K^T are key rows: contiguous in LDS
      v16bf b0 = load_b_tile(kbuf + n16 * D_SZ + d0, h);
      v16bf b1 = load_b_tile(kbuf + (16 + n16) * D_SZ + d0, h);
      c0 = WMMA_BF16(a, b0, c0);
      c1 = WMMA_BF16(a, b1, c1);
    }

    const int k0g = kt * 32 + n16;
    const int k1g = kt * 32 + 16 + n16;
    const bool mk0 = mbase[k0g] != 0;
    const bool mk1 = mbase[k1g] != 0;
#pragma unroll
    for (int r = 0; r < 8; ++r) {
      const int m  = r + 8 * h;              // C layout: vgpr r, half h -> row m
      const int qg = q0 + qrow0 + m;
      float p0 = (mk0 && (k0g != qg)) ? __expf(c0[r]) : 0.0f;  // |s| small: no max shift
      float p1 = (mk1 && (k1g != qg)) ? __expf(c1[r]) : 0.0f;
      psum[r] += p0 + p1;
      Pb[(qrow0 + m) * N_SZ + k0g] = (bf16_t)p0;
      Pb[(qrow0 + m) * N_SZ + k1g] = (bf16_t)p1;
    }
  }

#pragma unroll
  for (int r = 0; r < 8; ++r)
    atomicAdd(&rsum[qrow0 + r + 8 * h], psum[r]);   // ds_add_f32 reduction over 16 lanes
  __syncthreads();

  float rinv[8];
#pragma unroll
  for (int r = 0; r < 8; ++r) rinv[r] = 1.0f / rsum[qrow0 + r + 8 * h];

  // ---------------- Stage 2: O = P * V ----------------
  bf16_t* ab = attn + ((size_t)b * N_SZ + q0) * D_SZ;

  for (int nt = 0; nt < D_SZ; nt += 64) {
    v8f o0 = zero8(), o1 = zero8(), o2 = zero8(), o3 = zero8();
    for (int kt = 0; kt < N_SZ / 32; ++kt) {
      __syncthreads();
      {  // transpose V chunk [32 keys][64 dims] -> Vt[64 dims][32 keys]
        const int k  = tid & 31;
        const int dd = (tid >> 5) * 16;
        const bf16_t* src = xb + (size_t)(kt * 32 + k) * D_SZ + nt + dd;
        v8bf s0 = *(const v8bf*)src;
        v8bf s1 = *(const v8bf*)(src + 8);
#pragma unroll
        for (int i = 0; i < 8; ++i) {
          Vt[(dd + i) * 32 + k]     = s0[i];
          Vt[(dd + 8 + i) * 32 + k] = s1[i];
        }
      }
      __syncthreads();
      v16bf a = load_a_tile(Pb + (qrow0 + n16) * N_SZ + kt * 32, h);
      o0 = WMMA_BF16(a, load_b_tile(Vt + (0 * 16 + n16) * 32, h), o0);
      o1 = WMMA_BF16(a, load_b_tile(Vt + (1 * 16 + n16) * 32, h), o1);
      o2 = WMMA_BF16(a, load_b_tile(Vt + (2 * 16 + n16) * 32, h), o2);
      o3 = WMMA_BF16(a, load_b_tile(Vt + (3 * 16 + n16) * 32, h), o3);
    }
#pragma unroll
    for (int r = 0; r < 8; ++r) {
      const int m = r + 8 * h;
      bf16_t* dst = ab + (size_t)(qrow0 + m) * D_SZ + nt + n16;
      dst[0]  = (bf16_t)(o0[r] * rinv[r]);
      dst[16] = (bf16_t)(o1[r] * rinv[r]);
      dst[32] = (bf16_t)(o2[r] * rinv[r]);
      dst[48] = (bf16_t)(o3[r] * rinv[r]);
    }
  }
}

// ---------------------------------------------------------------------------
// Projection: out[m,n] = sum_k attn[m,k] * W[n,k] + bias[n]; 16 rows per wave.
// ---------------------------------------------------------------------------
__global__ void __launch_bounds__(256)
proj_kernel(const bf16_t* __restrict__ attn, const bf16_t* __restrict__ Wb,
            const float* __restrict__ bias, float* __restrict__ out) {
  const int tid  = threadIdx.x;
  const int wave = tid >> 5;
  const int lane = tid & 31;
  const int n16  = lane & 15;
  const int h    = lane >> 4;

  const size_t row0 = ((size_t)blockIdx.x * 8 + wave) * 16;
  const bf16_t* A = attn + row0 * D_SZ;
  float* O = out + row0 * D_SZ;

  for (int nt = 0; nt < D_SZ; nt += 64) {
    v8f acc0 = zero8(), acc1 = zero8(), acc2 = zero8(), acc3 = zero8();
#pragma unroll 4
    for (int k0 = 0; k0 < D_SZ; k0 += 32) {
      v16bf a = load_a_tile(A + (size_t)n16 * D_SZ + k0, h);
      acc0 = WMMA_BF16(a, load_b_tile(Wb + (size_t)(nt + 0 * 16 + n16) * D_SZ + k0, h), acc0);
      acc1 = WMMA_BF16(a, load_b_tile(Wb + (size_t)(nt + 1 * 16 + n16) * D_SZ + k0, h), acc1);
      acc2 = WMMA_BF16(a, load_b_tile(Wb + (size_t)(nt + 2 * 16 + n16) * D_SZ + k0, h), acc2);
      acc3 = WMMA_BF16(a, load_b_tile(Wb + (size_t)(nt + 3 * 16 + n16) * D_SZ + k0, h), acc3);
    }
    const float b0 = bias[nt + 0 * 16 + n16];
    const float b1 = bias[nt + 1 * 16 + n16];
    const float b2 = bias[nt + 2 * 16 + n16];
    const float b3 = bias[nt + 3 * 16 + n16];
#pragma unroll
    for (int r = 0; r < 8; ++r) {
      const int m = r + 8 * h;
      float* dst = O + (size_t)m * D_SZ + nt + n16;
      dst[0]  = acc0[r] + b0;
      dst[16] = acc1[r] + b1;
      dst[32] = acc2[r] + b2;
      dst[48] = acc3[r] + b3;
    }
  }
}

// ---------------------------------------------------------------------------
extern "C" void kernel_launch(void* const* d_in, const int* in_sizes, int n_in,
                              void* d_out, int out_size, void* d_ws, size_t ws_size,
                              hipStream_t stream) {
  (void)in_sizes; (void)n_in; (void)out_size; (void)ws_size;

  const float*         x    = (const float*)d_in[0];
  const unsigned char* mask = (const unsigned char*)d_in[1];
  const float*         W    = (const float*)d_in[2];
  const float*         bias = (const float*)d_in[3];
  float*               out  = (float*)d_out;

  // workspace: xs (64MB) | Wb (0.5MB) | attn (64MB), all bf16
  bf16_t* xs   = (bf16_t*)d_ws;
  bf16_t* Wb   = xs + (size_t)B_SZ * N_SZ * D_SZ;
  bf16_t* attn = Wb + (size_t)D_SZ * D_SZ;

  const float scale = 1.0f / sqrtf((float)D_SZ);

  {
    int n4 = (B_SZ * N_SZ * D_SZ) / 4;
    prep_kernel<<<(n4 + 255) / 256, 256, 0, stream>>>(x, xs, scale, n4);
  }
  {
    int n4 = (D_SZ * D_SZ) / 4;
    prep_kernel<<<(n4 + 255) / 256, 256, 0, stream>>>(W, Wb, 1.0f, n4);
  }

  size_t smem = (size_t)(64 * D_SZ + 2 * 32 * D_SZ + 64 * N_SZ + 64 * 32) * sizeof(bf16_t)
              + 64 * sizeof(float);   // ~260 KB dynamic LDS (WGP has 320 KB)
  attn_kernel<<<B_SZ * (N_SZ / 64), 128, smem, stream>>>(xs, mask, attn);

  proj_kernel<<<(B_SZ * N_SZ) / (16 * 8), 256, 0, stream>>>(attn, Wb, bias, out);
}
